// scaled_dot_product_attention_22256520528885
// MI455X (gfx1250) — matrix operand order repaired and checked by reading, not compiled, not run on hardware
//
#include <hip/hip_runtime.h>

typedef __attribute__((ext_vector_type(16))) _Float16 v16h;
typedef __attribute__((ext_vector_type(8)))  _Float16 v8h;
typedef __attribute__((ext_vector_type(4)))  _Float16 v4h;
typedef __attribute__((ext_vector_type(2)))  _Float16 v2h;
typedef __attribute__((ext_vector_type(8)))  float    v8f;
typedef __attribute__((ext_vector_type(4)))  float    v4f;

#define SDIM   2048
#define DHEAD  64
#define BM     128    // query rows per workgroup (8 waves x 16)
#define BN     64     // keys per LDS tile; softmax spans tile PAIRS (128 keys)
#define NW     8
#define LSTR   72     // padded K/V LDS row stride in halfs (144B)
#define PSTR   136    // padded P LDS row stride in halfs (272B, 128 cols + pad)

#define QSCALE 0.1803368801111204f   // 0.125 * log2(e); softmax in log2 domain
#define MASKVAL (-30000.0f)

#define DPP_QUAD_XOR1   0xB1
#define DPP_QUAD_XOR2   0x4E
#define DPP_HALF_MIRROR 0x141
#define DPP_ROW_MIRROR  0x140
#define DPP_MAX_STEP(x, ctrl)                                                  \
  (x) = fmaxf((x), __int_as_float(__builtin_amdgcn_update_dpp(                 \
            0, __float_as_int(x), (ctrl), 0xF, 0xF, true)))

static __device__ __forceinline__ v16h ld_frag(const _Float16* p, int second_off) {
  v8h a = *reinterpret_cast<const v8h*>(p);
  v8h b = *reinterpret_cast<const v8h*>(p + second_off);
  v16h r;
#pragma unroll
  for (int j = 0; j < 8; ++j) { r[j] = a[j]; r[j + 8] = b[j]; }
  return r;
}

// stage one 64x64 f32 tile of K and V into registers
static __device__ __forceinline__ void stage_tile(const float* kp, const float* vp,
                                                  v4f* kst, v4f* vst) {
#pragma unroll
  for (int i = 0; i < 4; ++i) kst[i] = *reinterpret_cast<const v4f*>(kp + i * 4);
  vst[0] = *reinterpret_cast<const v4f*>(vp);
  vst[1] = *reinterpret_cast<const v4f*>(vp + 4);
  vst[2] = *reinterpret_cast<const v4f*>(vp + DHEAD);
  vst[3] = *reinterpret_cast<const v4f*>(vp + DHEAD + 4);
}

// convert + store staged tile: K row-major, V transposed
static __device__ __forceinline__ void store_tile(_Float16* kbuf, _Float16* vbuf,
                                                  int krow, int kcb, int vrp, int vcg,
                                                  const v4f* kst, const v4f* vst) {
  _Float16* kd = &kbuf[krow * LSTR + kcb];
#pragma unroll
  for (int i = 0; i < 4; ++i) {
    v4h h; h[0] = (_Float16)kst[i][0]; h[1] = (_Float16)kst[i][1];
    h[2] = (_Float16)kst[i][2]; h[3] = (_Float16)kst[i][3];
    *reinterpret_cast<v4h*>(kd + i * 4) = h;
  }
#pragma unroll
  for (int c = 0; c < 8; ++c) {
    v2h h; h[0] = (_Float16)vst[c >> 2][c & 3];
    h[1] = (_Float16)vst[2 + (c >> 2)][c & 3];
    *reinterpret_cast<v2h*>(&vbuf[(vcg * 8 + c) * LSTR + vrp * 2]) = h;
  }
}

// QK^T over one 64-key tile -> 4 chunk accumulators (compile-time destinations)
static __device__ __forceinline__ void qk_tile(const _Float16* kbuf, const v16h* qfrag,
                                               int lo16, int hi, v8f* out) {
#pragma unroll
  for (int cn = 0; cn < 4; ++cn) {
    v8f acc;
#pragma unroll
    for (int i = 0; i < 8; ++i) acc[i] = 0.0f;
#pragma unroll
    for (int kkq = 0; kkq < 2; ++kkq) {
      const _Float16* bp = &kbuf[(16 * cn + lo16) * LSTR + kkq * 32 + hi * 8];
      v16h bfrag = ld_frag(bp, 16);
      acc = __builtin_amdgcn_wmma_f32_16x16x32_f16(
          false, qfrag[kkq], false, bfrag, (short)0, acc, false, false);
    }
    out[cn] = acc;
  }
}

__global__ __launch_bounds__(256) void fattn_f16wmma_kernel(
    const float* __restrict__ Q, const float* __restrict__ K,
    const float* __restrict__ V, float* __restrict__ O) {
  __shared__ _Float16 Ks_e[BN * LSTR];       // even tile K
  __shared__ _Float16 Ks_o[BN * LSTR];       // odd tile K
  __shared__ _Float16 Vt_o[DHEAD * LSTR];    // odd tile V (transposed)
  __shared__ _Float16 Vt_e[2][DHEAD * LSTR]; // even tile V, 2-deep rotation
  __shared__ _Float16 Ps[NW * 16 * PSTR];    // per-wave P staging, 128 cols

  const int tid  = threadIdx.x;
  const int wave = tid >> 5;
  const int lane = tid & 31;
  const int lo16 = lane & 15;
  const int hi   = lane >> 4;

  const int qb = blockIdx.x;
  const int bh = blockIdx.y;
  const size_t head_off = (size_t)bh * SDIM * DHEAD;
  const float* Qh = Q + head_off;
  const float* Kh = K + head_off;
  const float* Vh = V + head_off;
  float*       Oh = O + head_off;

  const int qrow_base = qb * BM + wave * 16;
  const bool upper_wave = (wave >= 4);   // rows in upper half of the 128-row block

  // ---- Q fragments (A-layout), pre-scaled into log2 domain ----
  v16h qfrag[2];
  {
    const float* qrow = Qh + (size_t)(qrow_base + lo16) * DHEAD;
#pragma unroll
    for (int kk = 0; kk < 2; ++kk) {
#pragma unroll
      for (int g = 0; g < 2; ++g) {
        const int d0 = kk * 32 + g * 16 + hi * 8;
#pragma unroll
        for (int j = 0; j < 8; ++j)
          qfrag[kk][g * 8 + j] = (_Float16)(qrow[d0 + j] * QSCALE);
      }
    }
  }

  v16h ones;
#pragma unroll
  for (int j = 0; j < 16; ++j) ones[j] = (_Float16)1.0f;

  v8f acc_o[4], acc_l, sc[8];
#pragma unroll
  for (int i = 0; i < 8; ++i) acc_l[i] = 0.0f;
#pragma unroll
  for (int c = 0; c < 4; ++c)
#pragma unroll
    for (int i = 0; i < 8; ++i) acc_o[c][i] = 0.0f;

  float mstate[8];
#pragma unroll
  for (int r = 0; r < 8; ++r) mstate[r] = -__builtin_inff();

  // ---- per-thread cooperative staging assignments ----
  const int krow = tid >> 2, kcb = (tid & 3) * 16;
  const int vrp  = tid >> 3, vcg = tid & 7;
  const float* kstage = Kh + (size_t)krow * DHEAD + kcb;
  const float* vstage = Vh + (size_t)(vrp * 2) * DHEAD + vcg * 8;

  v4f kst[4], vst[4];
  // prologue: tile 0 -> Ks_e, Vt_e[0]
  stage_tile(kstage, vstage, kst, vst);
  kstage += BN * DHEAD; vstage += BN * DHEAD;
  store_tile(Ks_e, Vt_e[0], krow, kcb, vrp, vcg, kst, vst);

  for (int pb = 0; pb <= qb; ++pb) {
    const bool lastpair = (pb == qb);
    const int  vbe = pb & 1;

    // ======== segment A: tile 2pb (even) ========
    stage_tile(kstage, vstage, kst, vst);          // tile 2pb+1
    kstage += BN * DHEAD; vstage += BN * DHEAD;

    __syncthreads();                               // tile 2pb resident

    qk_tile(Ks_e, qfrag, lo16, hi, &sc[0]);        // -> sc[0..3]

    store_tile(Ks_o, Vt_o, krow, kcb, vrp, vcg, kst, vst);  // tile 2pb+1

    // ======== segment B: tile 2pb+1 (odd) + flush ========
    if (!lastpair) {
      stage_tile(kstage, vstage, kst, vst);        // tile 2pb+2
      kstage += BN * DHEAD; vstage += BN * DHEAD;
    }

    __syncthreads();                               // tile 2pb+1 resident

    const bool upper = upper_wave || !lastpair;    // odd tile needed by this wave?
    if (upper) {
      qk_tile(Ks_o, qfrag, lo16, hi, &sc[4]);      // -> sc[4..7]
    } else {
#pragma unroll
      for (int cn = 4; cn < 8; ++cn)
#pragma unroll
        for (int i = 0; i < 8; ++i) sc[cn][i] = MASKVAL;
    }

    // ---- softmax over 128 keys (log2 domain) ----
    const int kstart0 = pb * 2 * BN;
#pragma unroll
    for (int r = 0; r < 8; ++r) {
      const int qrow = qrow_base + r + 8 * hi;
      if (lastpair) {
        const int kcol0 = kstart0 + lo16;
#pragma unroll
        for (int cn = 0; cn < 8; ++cn)
          if (kcol0 + 16 * cn > qrow) sc[cn][r] = MASKVAL;
      }
      float mloc =
          fmaxf(fmaxf(fmaxf(sc[0][r], sc[1][r]), fmaxf(sc[2][r], sc[3][r])),
                fmaxf(fmaxf(sc[4][r], sc[5][r]), fmaxf(sc[6][r], sc[7][r])));
      DPP_MAX_STEP(mloc, DPP_QUAD_XOR1);
      DPP_MAX_STEP(mloc, DPP_QUAD_XOR2);
      DPP_MAX_STEP(mloc, DPP_HALF_MIRROR);
      DPP_MAX_STEP(mloc, DPP_ROW_MIRROR);
      const float mnew = fmaxf(mstate[r], mloc);
      const float corr = __builtin_amdgcn_exp2f(mstate[r] - mnew);
      mstate[r] = mnew;
      _Float16* prow = &Ps[(wave * 16 + r + 8 * hi) * PSTR + lo16];
#pragma unroll
      for (int cn = 0; cn < 8; ++cn) {
        if (cn < 4 || upper) {
          const float p = __builtin_amdgcn_exp2f(sc[cn][r] - mnew);
          prow[16 * cn] = (_Float16)p;
        }
      }
      acc_l[r] *= corr;
#pragma unroll
      for (int c = 0; c < 4; ++c) acc_o[c][r] *= corr;
    }

    // ---- O += P*V, l += P*ones over 128 keys ----
#pragma unroll
    for (int kk = 0; kk < 4; ++kk) {
      if (kk < 2 || upper) {
        const _Float16* vb = (kk < 2) ? Vt_e[vbe] : Vt_o;
        v16h pfrag = ld_frag(&Ps[(wave * 16 + lo16) * PSTR + kk * 32 + hi * 8], 16);
        acc_l = __builtin_amdgcn_wmma_f32_16x16x32_f16(
            false, pfrag, false, ones, (short)0, acc_l, false, false);
#pragma unroll
        for (int c = 0; c < 4; ++c) {
          v16h vfrag = ld_frag(&vb[(16 * c + lo16) * LSTR + (kk & 1) * 32 + hi * 8], 16);
          acc_o[c] = __builtin_amdgcn_wmma_f32_16x16x32_f16(
              false, pfrag, false, vfrag, (short)0, acc_o[c], false, false);
        }
      }
    }

    if (!lastpair) {
      store_tile(Ks_e, Vt_e[vbe ^ 1], krow, kcb, vrp, vcg, kst, vst);  // tile 2pb+2
    }
  }

  // ---- normalize and write O ----
#pragma unroll
  for (int r = 0; r < 8; ++r) {
    const float inv = __builtin_amdgcn_rcpf(acc_l[r]);
    float* orow = Oh + (size_t)(qrow_base + r + 8 * hi) * DHEAD;
#pragma unroll
    for (int c = 0; c < 4; ++c)
      orow[16 * c + lo16] = acc_o[c][r] * inv;
  }
}

extern "C" void kernel_launch(void* const* d_in, const int* in_sizes, int n_in,
                              void* d_out, int out_size, void* d_ws, size_t ws_size,
                              hipStream_t stream) {
  const float* Q = (const float*)d_in[0];
  const float* K = (const float*)d_in[1];
  const float* V = (const float*)d_in[2];
  float* O = (float*)d_out;
  const int BH = in_sizes[0] / (SDIM * DHEAD);  // B*H = 32
  dim3 grid(SDIM / BM, BH);
  hipLaunchKernelGGL(fattn_f16wmma_kernel, grid, dim3(256), 0, stream, Q, K, V, O);
}
